// Model_40785009443559
// MI455X (gfx1250) — matrix operand order repaired and checked by reading, not compiled, not run on hardware
//
#include <hip/hip_runtime.h>
#include <hip/hip_bf16.h>
#include <math.h>

// ---------------- model constants ----------------
static constexpr int kBS = 32, kSEQ = 512, kNV = 32;
static constexpr int kPATCH = 16, kSTRIDE = 8;
static constexpr int kD = 512, kNH = 8, kL = 4, kHD = 64, kHID = 1536;
static constexpr int kPNUM = 64, kPRED = 96;
static constexpr int kT = kBS * kPNUM * kNV;          // 65536 tokens in both stages
static constexpr float kEPS = 1e-5f;

typedef __bf16 bf16_t;
typedef __attribute__((ext_vector_type(16))) __bf16 v16bf;
typedef __attribute__((ext_vector_type(8)))  __bf16 v8bf;
typedef __attribute__((ext_vector_type(8)))  float  v8f;
typedef __attribute__((ext_vector_type(4)))  unsigned int u32x4;
typedef __attribute__((ext_vector_type(8)))  int i32x8;
typedef __attribute__((ext_vector_type(4)))  int i32x4;

// ---------------- WMMA helpers ----------------
// CDNA5 wave32 16-bit A fragment (16x32, row-major source, leading dim ld):
//   lane m (0..15)/m+16 holds K = {h*8..h*8+7} U {16+h*8..16+h*8+7}, h = lane>>4
__device__ __forceinline__ v16bf load_frag_a(const bf16_t* __restrict__ base, int ld) {
  int lane = threadIdx.x & 31;
  int hlf = lane >> 4, m = lane & 15;
  const bf16_t* p = base + (size_t)m * ld + hlf * 8;
  v8bf lo = *reinterpret_cast<const v8bf*>(p);
  v8bf hi = *reinterpret_cast<const v8bf*>(p + 16);
  v16bf f;
#pragma unroll
  for (int e = 0; e < 8; ++e) { f[e] = lo[e]; f[e + 8] = hi[e]; }
  return f;
}

// B fragment (32x16) loaded from an N-major ("transposed") matrix Bt[N][K]:
//   lane n (0..15)/n+16 holds K = {h*8..h*8+7} U {16+h*8..16+h*8+7} for column n.
__device__ __forceinline__ v16bf load_frag_b(const bf16_t* __restrict__ baseT, int ld) {
  int lane = threadIdx.x & 31;
  int hlf = lane >> 4, n = lane & 15;
  const bf16_t* p = baseT + (size_t)n * ld + hlf * 8;
  v8bf lo = *reinterpret_cast<const v8bf*>(p);
  v8bf hi = *reinterpret_cast<const v8bf*>(p + 16);
  v16bf f;
#pragma unroll
  for (int e = 0; e < 8; ++e) { f[e] = lo[e]; f[e + 8] = hi[e]; }
  return f;
}

__device__ __forceinline__ v8f wmma_bf16(v16bf a, v16bf b, v8f c) {
  return __builtin_amdgcn_wmma_f32_16x16x32_bf16(false, a, false, b, (short)0, c, false, false);
}

// ---------------- TDM: 2D bf16 tile global -> LDS (amdgpu-toolchain 6-arg builtin) ------
// D# per CDNA5 ISA. 2D tensor, data_size=1 (2 bytes),
// tile = rows x rowlen elements, global row stride = gstride elements.
__device__ __forceinline__ void tdm_load_2d_bf16(unsigned lds_off, const void* gptr,
                                                 int rows, int rowlen, int gstride) {
  unsigned long long ga = (unsigned long long)(uintptr_t)gptr;
  u32x4 g0;
  g0[0] = 1u;                                            // count=1, user descriptor
  g0[1] = lds_off;                                       // lds_addr[31:0]
  g0[2] = (unsigned)(ga & 0xffffffffu);                  // global_addr[31:0]
  g0[3] = (unsigned)((ga >> 32) & 0x1ffffffu) | (2u << 30); // global_addr[56:32], type=2
  i32x8 g1;
  g1[0] = (1 << 16);                                     // wg_mask=0, data_size=1 (2B)
  g1[1] = (rowlen & 0xffff) << 16;                       // tensor_dim0[15:0] @bits63:48
  g1[2] = ((unsigned)rowlen >> 16) | ((rows & 0xffff) << 16); // dim0 hi | tensor_dim1 lo
  g1[3] = ((unsigned)rows >> 16) | ((rowlen & 0xffff) << 16); // dim1 hi | tile_dim0
  g1[4] = (rows & 0xffff);                               // tile_dim1 | tile_dim2=0
  g1[5] = gstride;                                       // tensor_dim0_stride[31:0]
  g1[6] = 0;                                             // stride hi | dim1_stride lo
  g1[7] = 0;
  i32x4 z4 = {0, 0, 0, 0};
  i32x8 z8 = {0, 0, 0, 0, 0, 0, 0, 0};
  __builtin_amdgcn_tensor_load_to_lds(g0, g1, z4, z4, z8, 0);
}

// ---------------- generic GEMM: C[M,N] = A[M,K](bf16) x Bt[N,K](bf16) ----------------
#define GF_BIAS     1
#define GF_RESID    2
#define GF_OUTF32   4
#define GF_OUTBF16  8
#define GF_SILUGATE 16

__global__ __launch_bounds__(256) void gemm_bf16_kernel(
    const bf16_t* __restrict__ A, const bf16_t* __restrict__ Bt,
    int M, int N, int K,
    const float* __restrict__ bias, const float* __restrict__ resid,
    const bf16_t* __restrict__ gate,
    float* __restrict__ Cf, bf16_t* __restrict__ Cb, int flags) {
  int tid = threadIdx.x;
  int w = tid >> 5, lane = tid & 31;
  int wm = w & 1, wn = w >> 1;                  // 2x4 wave grid -> 64x128 block tile
  int row0 = blockIdx.y * 64 + wm * 32;
  int col0 = blockIdx.x * 128 + wn * 32;
  if (row0 + 32 > M || col0 + 32 > N) return;   // uniform per wave

  const bf16_t* A0 = A  + (size_t)row0 * K;
  const bf16_t* A1 = A  + (size_t)(row0 + 16) * K;
  const bf16_t* B0 = Bt + (size_t)col0 * K;
  const bf16_t* B1 = Bt + (size_t)(col0 + 16) * K;

  v8f acc[2][2] = {};
  for (int k0 = 0; k0 < K; k0 += 32) {
    __builtin_prefetch(A0 + k0 + 256, 0, 0);    // global_prefetch_b8
    __builtin_prefetch(B0 + k0 + 256, 0, 0);
    v16bf a0 = load_frag_a(A0 + k0, K);
    v16bf a1 = load_frag_a(A1 + k0, K);
    v16bf b0 = load_frag_b(B0 + k0, K);
    v16bf b1 = load_frag_b(B1 + k0, K);
    acc[0][0] = wmma_bf16(a0, b0, acc[0][0]);
    acc[0][1] = wmma_bf16(a0, b1, acc[0][1]);
    acc[1][0] = wmma_bf16(a1, b0, acc[1][0]);
    acc[1][1] = wmma_bf16(a1, b1, acc[1][1]);
  }

  int hlf = lane >> 4, nn = lane & 15;
#pragma unroll
  for (int tm = 0; tm < 2; ++tm) {
#pragma unroll
    for (int tn = 0; tn < 2; ++tn) {
#pragma unroll
      for (int j = 0; j < 8; ++j) {
        int r = row0 + tm * 16 + hlf * 8 + j;
        int c = col0 + tn * 16 + nn;
        float val = acc[tm][tn][j];
        size_t idx = (size_t)r * N + c;
        if (flags & GF_BIAS)     val += bias[c];
        if (flags & GF_SILUGATE) { float g = (float)gate[idx]; val *= g / (1.f + __expf(-g)); }
        if (flags & GF_RESID)    val += resid[idx];
        if (flags & GF_OUTF32)   Cf[idx] = val;
        if (flags & GF_OUTBF16)  Cb[idx] = (bf16_t)val;
      }
    }
  }
}

// ---------------- weight convert: f32 W[Kin,N] -> bf16 Wt[N,Kpad] (zero pad K) -----------
__global__ __launch_bounds__(256) void transpose_pad_kernel(
    const float* __restrict__ W, bf16_t* __restrict__ Wt, int Kin, int Kpad, int N) {
  size_t i = (size_t)blockIdx.x * blockDim.x + threadIdx.x;
  size_t total = (size_t)N * Kpad;
  if (i >= total) return;
  int n = (int)(i / Kpad), k = (int)(i % Kpad);
  float v = (k < Kin) ? W[(size_t)k * N + n] : 0.f;
  Wt[i] = (bf16_t)v;
}

// ---------------- RMS norm (wave per row), optional rel->enc permute --------------------
__global__ __launch_bounds__(256) void rmsnorm_kernel(
    const float* __restrict__ X, const float* __restrict__ w,
    bf16_t* __restrict__ outb, float* __restrict__ outf, int T, int permute) {
  int wid = threadIdx.x >> 5, lane = threadIdx.x & 31;
  int row = blockIdx.x * 8 + wid;
  if (row >= T) return;
  const float* x = X + (size_t)row * kD;
  float ss = 0.f;
  for (int i = lane; i < kD; i += 32) { float t = x[i]; ss += t * t; }
#pragma unroll
  for (int off = 16; off; off >>= 1) ss += __shfl_xor(ss, off, 32);
  float r = rsqrtf(ss * (1.f / kD) + kEPS);
  int orow = row;
  if (permute) {  // in row = (b*PNUM+p)*NV+v  ->  out row = (b*NV+v)*PNUM+p
    int b = row / (kPNUM * kNV);
    int rem = row % (kPNUM * kNV);
    int p = rem / kNV, v = rem % kNV;
    orow = (b * kNV + v) * kPNUM + p;
  }
  for (int i = lane; i < kD; i += 32) {
    float val = x[i] * r * w[i];
    if (outb) outb[(size_t)orow * kD + i] = (bf16_t)val;
    if (outf) outf[(size_t)orow * kD + i] = val;
  }
}

// ---------------- RoPE (enc stage), in place on q and k ---------------------------------
__global__ __launch_bounds__(256) void rope_kernel(
    bf16_t* __restrict__ q, bf16_t* __restrict__ k, int T) {
  int t = blockIdx.x;
  int pair = threadIdx.x;                // 256 == D/2
  int h = pair / (kHD / 2), i = pair % (kHD / 2);
  int pos = t % kPNUM;
  float ang = (float)pos * __powf(10000.f, -2.f * (float)i / (float)kHD);
  float s, c;
  __sincosf(ang, &s, &c);
  size_t base = (size_t)t * kD + h * kHD + 2 * i;
  float qr = (float)q[base], qi = (float)q[base + 1];
  q[base]     = (bf16_t)(qr * c - qi * s);
  q[base + 1] = (bf16_t)(qr * s + qi * c);
  float kr = (float)k[base], ki = (float)k[base + 1];
  k[base]     = (bf16_t)(kr * c - ki * s);
  k[base + 1] = (bf16_t)(kr * s + ki * c);
}

// ---------------- attention: one workgroup per (batch, head) ----------------------------
__global__ __launch_bounds__(256) void attention_kernel(
    const bf16_t* __restrict__ q, const bf16_t* __restrict__ k, const bf16_t* __restrict__ v,
    bf16_t* __restrict__ o, int S, int causal) {
  __shared__ __align__(16) bf16_t Qs[64 * kHD];
  __shared__ __align__(16) bf16_t Ks[64 * kHD];
  __shared__ __align__(16) bf16_t Vt[kHD * 64];   // transposed V
  __shared__ __align__(16) float  Sc[64 * 64];
  __shared__ __align__(16) bf16_t Ps[64 * 64];

  int bh = blockIdx.x;
  int batch = bh / kNH, head = bh % kNH;
  int tid = threadIdx.x, w = tid >> 5, lane = tid & 31;
  int hlf = lane >> 4, nn = lane & 15;

  // Q and K tiles (S x HD bf16, row stride D) via Tensor Data Mover, issued by wave 0.
  size_t goff = ((size_t)(batch * S)) * kD + head * kHD;
  if (w == 0) {
    tdm_load_2d_bf16((unsigned)(uintptr_t)Qs, q + goff, S, kHD, kD);
    tdm_load_2d_bf16((unsigned)(uintptr_t)Ks, k + goff, S, kHD, kD);
    __builtin_amdgcn_s_wait_tensorcnt(0);
  }
  // V needs a transpose -> cooperative load (TDM cannot transpose).
  for (int idx = tid; idx < S * kHD; idx += 256) {
    int s = idx / kHD, d = idx % kHD;
    Vt[d * S + s] = v[goff + (size_t)s * kD + d];
  }
  __syncthreads();

  int st = S >> 4;
  const float scale = 0.125f;                     // 1/sqrt(64)
  for (int tile = w; tile < st * st; tile += 8) { // scores = Q K^T
    int tm = tile / st, tn = tile % st;
    v8f acc = {};
    for (int k0 = 0; k0 < kHD; k0 += 32) {
      v16bf a = load_frag_a(Qs + tm * 16 * kHD + k0, kHD);
      v16bf b = load_frag_b(Ks + tn * 16 * kHD + k0, kHD);
      acc = wmma_bf16(a, b, acc);
    }
#pragma unroll
    for (int j = 0; j < 8; ++j) {
      int r = tm * 16 + hlf * 8 + j;
      int c = tn * 16 + nn;
      float val = acc[j] * scale;
      if (causal && c > r) val = -1e30f;
      Sc[r * S + c] = val;
    }
  }
  __syncthreads();

  for (int r = tid; r < S; r += 256) {            // softmax rows
    float mx = -1e30f;
    for (int c = 0; c < S; ++c) mx = fmaxf(mx, Sc[r * S + c]);
    float sum = 0.f;
    for (int c = 0; c < S; ++c) { float e = __expf(Sc[r * S + c] - mx); Sc[r * S + c] = e; sum += e; }
    float inv = 1.f / sum;
    for (int c = 0; c < S; ++c) Ps[r * S + c] = (bf16_t)(Sc[r * S + c] * inv);
  }
  __syncthreads();

  for (int tile = w; tile < st * (kHD / 16); tile += 8) {  // O = P V
    int tm = tile / (kHD / 16), tn = tile % (kHD / 16);
    v8f acc = {};
    for (int k0 = 0; k0 < S; k0 += 32) {
      v16bf a = load_frag_a(Ps + tm * 16 * S + k0, S);
      v16bf b = load_frag_b(Vt + tn * 16 * S + k0, S);
      acc = wmma_bf16(a, b, acc);
    }
#pragma unroll
    for (int j = 0; j < 8; ++j) {
      int r = tm * 16 + hlf * 8 + j;
      int c = tn * 16 + nn;
      o[((size_t)(batch * S + r)) * kD + head * kHD + c] = (bf16_t)acc[j];
    }
  }
}

// ---------------- instance norm + patching -> bf16 patches in (b,p,v) row order ---------
__global__ __launch_bounds__(256) void prep_kernel(
    const float* __restrict__ tokens, bf16_t* __restrict__ patches,
    float* __restrict__ mean_out, float* __restrict__ std_out) {
  int wid = threadIdx.x >> 5, lane = threadIdx.x & 31;
  int pair = blockIdx.x * 8 + wid;
  if (pair >= kBS * kNV) return;
  int b = pair / kNV, v = pair % kNV;
  const float* tb = tokens + (size_t)b * kSEQ * kNV + v;   // stride kNV over SEQ
  float sum = 0.f;
  for (int s = lane; s < kSEQ; s += 32) sum += tb[(size_t)s * kNV];
#pragma unroll
  for (int off = 16; off; off >>= 1) sum += __shfl_xor(sum, off, 32);
  float mean = sum / kSEQ;
  float var = 0.f;
  for (int s = lane; s < kSEQ; s += 32) { float d = tb[(size_t)s * kNV] - mean; var += d * d; }
#pragma unroll
  for (int off = 16; off; off >>= 1) var += __shfl_xor(var, off, 32);
  float stdev = sqrtf(var / kSEQ + kEPS);
  if (lane == 0) { mean_out[pair] = mean; std_out[pair] = stdev; }
  float inv = 1.f / stdev;
  for (int p = 0; p < kPNUM; ++p) {
    float val = 0.f;                       // K padded 16 -> 32 with zeros
    if (lane < kPATCH) {
      int sidx = p * kSTRIDE + lane;
      if (sidx >= kSEQ) sidx = kSEQ - 1;   // repeat-last padding
      val = (tb[(size_t)sidx * kNV] - mean) * inv;
    }
    patches[((size_t)((b * kPNUM + p) * kNV + v)) * 32 + lane] = (bf16_t)val;
  }
}

// ---------------- z (BS*NV, PRED) -> out (BS, PRED, NV) with de-norm ---------------------
__global__ __launch_bounds__(256) void final_kernel(
    const float* __restrict__ z, const float* __restrict__ mean,
    const float* __restrict__ stdev, float* __restrict__ out) {
  int i = blockIdx.x * blockDim.x + threadIdx.x;
  if (i >= kBS * kPRED * kNV) return;
  int v = i % kNV;
  int t = (i / kNV) % kPRED;
  int b = i / (kNV * kPRED);
  int pair = b * kNV + v;
  out[i] = z[(size_t)pair * kPRED + t] * stdev[pair] + mean[pair];
}

// ======================== host orchestration ========================
extern "C" void kernel_launch(void* const* d_in, const int* in_sizes, int n_in,
                              void* d_out, int out_size, void* d_ws, size_t ws_size,
                              hipStream_t stream) {
  (void)in_sizes; (void)n_in; (void)out_size; (void)ws_size;

  const float* tokens = (const float*)d_in[0];
  const float* pe_w   = (const float*)d_in[1];
  const float* pe_b   = (const float*)d_in[2];
  const float* norm_w = (const float*)d_in[3];
  const float* head_w = (const float*)d_in[4];
  const float* head_b = (const float*)d_in[5];
  const float* rel_w[9]; const float* enc_w[9];
  for (int i = 0; i < 9; ++i) { rel_w[i] = (const float*)d_in[6 + i]; enc_w[i] = (const float*)d_in[15 + i]; }
  // index: 0 wq, 1 wk, 2 wv, 3 wo, 4 w1, 5 w2, 6 w3, 7 an, 8 fn (each stacked L deep)

  char* ws = (char*)d_ws;
  auto bump = [&](size_t bytes) -> char* {
    char* r = ws; ws += (bytes + 255) & ~(size_t)255; return r;
  };

  const size_t TD = (size_t)kT * kD;
  bf16_t* patches = (bf16_t*)bump((size_t)kT * 32 * sizeof(bf16_t));
  float*  meanb   = (float*)bump((size_t)kBS * kNV * sizeof(float));
  float*  stdb    = (float*)bump((size_t)kBS * kNV * sizeof(float));
  float*  zbuf    = (float*)bump((size_t)kBS * kNV * kPRED * sizeof(float));
  bf16_t* peT     = (bf16_t*)bump((size_t)kD * 32 * sizeof(bf16_t));
  bf16_t* headT   = (bf16_t*)bump((size_t)kPRED * kPNUM * kD * sizeof(bf16_t));

  bf16_t* Wt[2][kL][7];
  const size_t wsz[7] = {(size_t)kD * kD, (size_t)kD * kD, (size_t)kD * kD, (size_t)kD * kD,
                         (size_t)kHID * kD, (size_t)kD * kHID, (size_t)kHID * kD};
  for (int s = 0; s < 2; ++s)
    for (int l = 0; l < kL; ++l)
      for (int j = 0; j < 7; ++j)
        Wt[s][l][j] = (bf16_t*)bump(wsz[j] * sizeof(bf16_t));

  float*  H  = (float*)bump(TD * sizeof(float));     // rel residual stream
  float*  H2 = (float*)bump(TD * sizeof(float));     // enc residual stream (permuted)
  bf16_t* xn = (bf16_t*)bump(TD * sizeof(bf16_t));   // normed activations

  // phase-shared region: attention {q,k,v,ao} aliases ffn {g1,g}
  char* R = bump(2 * (size_t)kT * kHID * sizeof(bf16_t));
  bf16_t* qb = (bf16_t*)R;
  bf16_t* kb = qb + TD;
  bf16_t* vb = kb + TD;
  bf16_t* ao = vb + TD;
  bf16_t* g1 = (bf16_t*)R;
  bf16_t* gg = g1 + (size_t)kT * kHID;

  auto conv = [&](const float* W, bf16_t* Wtp, int Kin, int Kpad, int N) {
    size_t total = (size_t)N * Kpad;
    transpose_pad_kernel<<<(unsigned)((total + 255) / 256), 256, 0, stream>>>(W, Wtp, Kin, Kpad, N);
  };
  auto gemm = [&](const bf16_t* A, const bf16_t* Bt, int M, int N, int K,
                  const float* bias, const float* resid, const bf16_t* gate,
                  float* Cf, bf16_t* Cb, int flags) {
    dim3 grid((N + 127) / 128, (M + 63) / 64);
    gemm_bf16_kernel<<<grid, 256, 0, stream>>>(A, Bt, M, N, K, bias, resid, gate, Cf, Cb, flags);
  };

  // -------- weight conversion (every call; deterministic) --------
  conv(pe_w, peT, kPATCH, 32, kD);
  conv(head_w, headT, kPNUM * kD, kPNUM * kD, kPRED);
  for (int s = 0; s < 2; ++s) {
    const float* const* src = s ? enc_w : rel_w;
    for (int l = 0; l < kL; ++l) {
      conv(src[0] + (size_t)l * kD * kD,   Wt[s][l][0], kD,   kD,   kD);
      conv(src[1] + (size_t)l * kD * kD,   Wt[s][l][1], kD,   kD,   kD);
      conv(src[2] + (size_t)l * kD * kD,   Wt[s][l][2], kD,   kD,   kD);
      conv(src[3] + (size_t)l * kD * kD,   Wt[s][l][3], kD,   kD,   kD);
      conv(src[4] + (size_t)l * kD * kHID, Wt[s][l][4], kD,   kD,   kHID);
      conv(src[5] + (size_t)l * kHID * kD, Wt[s][l][5], kHID, kHID, kD);
      conv(src[6] + (size_t)l * kD * kHID, Wt[s][l][6], kD,   kD,   kHID);
    }
  }

  // -------- prep + patch embed --------
  prep_kernel<<<(kBS * kNV + 7) / 8, 256, 0, stream>>>(tokens, patches, meanb, stdb);
  gemm(patches, peT, kT, kD, 32, pe_b, nullptr, nullptr, H, nullptr, GF_BIAS | GF_OUTF32);

  // -------- transformer blocks --------
  auto run_block = [&](float* Hs, int stage, int l, int S, int causal, int rope) {
    const float* an = (stage ? enc_w : rel_w)[7] + (size_t)l * kD;
    const float* fn = (stage ? enc_w : rel_w)[8] + (size_t)l * kD;
    bf16_t* const* W = Wt[stage][l];
    rmsnorm_kernel<<<kT / 8, 256, 0, stream>>>(Hs, an, xn, nullptr, kT, 0);
    gemm(xn, W[0], kT, kD, kD, nullptr, nullptr, nullptr, nullptr, qb, GF_OUTBF16);
    gemm(xn, W[1], kT, kD, kD, nullptr, nullptr, nullptr, nullptr, kb, GF_OUTBF16);
    gemm(xn, W[2], kT, kD, kD, nullptr, nullptr, nullptr, nullptr, vb, GF_OUTBF16);
    if (rope) rope_kernel<<<kT, 256, 0, stream>>>(qb, kb, kT);
    attention_kernel<<<(kT / S) * kNH, 256, 0, stream>>>(qb, kb, vb, ao, S, causal);
    gemm(ao, W[3], kT, kD, kD, nullptr, Hs, nullptr, Hs, nullptr, GF_RESID | GF_OUTF32);
    rmsnorm_kernel<<<kT / 8, 256, 0, stream>>>(Hs, fn, xn, nullptr, kT, 0);
    gemm(xn, W[4], kT, kHID, kD, nullptr, nullptr, nullptr, nullptr, g1, GF_OUTBF16);
    gemm(xn, W[6], kT, kHID, kD, nullptr, nullptr, g1, nullptr, gg, GF_OUTBF16 | GF_SILUGATE);
    gemm(gg, W[5], kT, kD, kHID, nullptr, Hs, nullptr, Hs, nullptr, GF_RESID | GF_OUTF32);
  };

  for (int l = 0; l < kL; ++l) run_block(H, 0, l, kNV, 0, 0);           // rel: seq 32
  rmsnorm_kernel<<<kT / 8, 256, 0, stream>>>(H, norm_w, nullptr, H2, kT, 1);  // permute
  for (int l = 0; l < kL; ++l) run_block(H2, 1, l, kPNUM, 1, 1);        // enc: seq 64, causal+rope

  // -------- head + output --------
  rmsnorm_kernel<<<kT / 8, 256, 0, stream>>>(H2, norm_w, xn, nullptr, kT, 0);
  gemm(xn, headT, kBS * kNV, kPRED, kPNUM * kD, head_b, nullptr, nullptr,
       zbuf, nullptr, GF_BIAS | GF_OUTF32);
  final_kernel<<<(kBS * kPRED * kNV + 255) / 256, 256, 0, stream>>>(
      zbuf, meanb, stdb, (float*)d_out);
}
